// SpGAT_58265526338330
// MI455X (gfx1250) — matrix operand order repaired and checked by reading, not compiled, not run on hardware
//
#include <hip/hip_runtime.h>
#include <hip/hip_bf16.h>

typedef __attribute__((ext_vector_type(16))) _Float16 v16h;
typedef __attribute__((ext_vector_type(8)))  float    v8f;

#define ALPHA 0.2f
#define EPS 1e-16f

// ---------------------------------------------------------------------------
// Fragment loaders per CDNA5 ISA 7.12.2 (wave32 layouts)
// A: 16x32 f16 (MxK). lane m = l&15, half hi = l>>4.
//    VGPR q holds K = (q<4?0:16) + hi*8 + (q&3)*2  (pair).
// B: 32x16 f16 (KxN), loaded from Bt[Ncols][K] (pre-transposed).
//    lane n = l&15, hi = l>>4. VGPR q holds K = (q<4?0:16) + hi*8 + (q&3)*2 (pair).
// C/D: 16x16 f32, VGPR r: M = r + hi*8, N = l&15.
// ---------------------------------------------------------------------------
__device__ __forceinline__ v16h load_a_frag(const _Float16* A, int row0, int kb,
                                            int K, int lane) {
  int m = lane & 15, hi = lane >> 4;
  union { v16h v; unsigned u[8]; } f;
  const _Float16* base = A + (size_t)(row0 + m) * K + kb;
#pragma unroll
  for (int q = 0; q < 8; ++q) {
    int k0 = ((q < 4) ? 0 : 16) + hi * 8 + (q & 3) * 2;
    f.u[q] = *reinterpret_cast<const unsigned*>(base + k0);
  }
  return f.v;
}

__device__ __forceinline__ v16h load_b_frag(const _Float16* Bt /*[Nc][K]*/,
                                            int col0, int kb, int K, int lane) {
  int n = lane & 15, hi = lane >> 4;
  union { v16h v; unsigned u[8]; } f;
  const _Float16* base = Bt + (size_t)(col0 + n) * K + kb;
#pragma unroll
  for (int q = 0; q < 8; ++q)
    f.u[q] = *reinterpret_cast<const unsigned*>(base + ((q < 4) ? 0 : 16) + hi * 8 + (q & 3) * 2);
  return f.v;
}

// C[Nrows,Nc] = A[Nrows,K] * Bt^T   (one wave per 16-row strip)
__global__ void gat_wmma_gemm(const _Float16* __restrict__ A,
                              const _Float16* __restrict__ Bt,
                              float* __restrict__ C,
                              int Nrows, int K, int Nc) {
  int wave = (int)((blockIdx.x * (size_t)blockDim.x + threadIdx.x) >> 5);
  int lane = threadIdx.x & 31;
  int row0 = wave * 16;
  if (row0 >= Nrows) return;               // wave-uniform: EXEC stays all-ones
  int m = lane & 15, hi = lane >> 4;
  for (int col0 = 0; col0 < Nc; col0 += 16) {
    v8f acc = {};
    for (int kb = 0; kb < K; kb += 32) {
      v16h a = load_a_frag(A, row0, kb, K, lane);
      v16h b = load_b_frag(Bt, col0, kb, K, lane);
      acc = __builtin_amdgcn_wmma_f32_16x16x32_f16(
          false, a, false, b, (short)0, acc, false, false);
    }
#pragma unroll
    for (int r = 0; r < 8; ++r)
      C[(size_t)(row0 + r + hi * 8) * Nc + col0 + m] = acc[r];
  }
}

// ------------------------- format converters -------------------------------
__global__ void conv_f32_f16(const float* __restrict__ in,
                             _Float16* __restrict__ out, long long n) {
  long long i = (long long)blockIdx.x * blockDim.x + threadIdx.x;
  if (i < n) out[i] = (_Float16)in[i];
}

// W1[8][128][64] -> Bt1[512][128] ; Bt1[head*64+j][k] = W1[head][k][j]
__global__ void conv_W1(const float* __restrict__ W1, _Float16* __restrict__ Bt) {
  int i = blockIdx.x * blockDim.x + threadIdx.x;     // 512*128
  if (i >= 512 * 128) return;
  int col = i >> 7, k = i & 127;
  int head = col >> 6, j = col & 63;
  Bt[i] = (_Float16)W1[head * 8192 + k * 64 + j];
}

// W2[512][128] -> Bt2[128][512] ; Bt2[c][k] = W2[k][c]
__global__ void conv_W2(const float* __restrict__ W2, _Float16* __restrict__ Bt) {
  int i = blockIdx.x * blockDim.x + threadIdx.x;     // 128*512
  if (i >= 128 * 512) return;
  int col = i >> 9, k = i & 511;
  Bt[i] = (_Float16)W2[k * 128 + col];
}

// ----------------------- attention score kernels ---------------------------
// one wave per (node, head): s_src = h.a[:64], s_dst = h.a[64:128]
__global__ void scores1_kernel(const float* __restrict__ h1,
                               const float* __restrict__ a1,
                               float* __restrict__ ssrc, float* __restrict__ sdst,
                               int N) {
  long long wave = ((long long)blockIdx.x * blockDim.x + threadIdx.x) >> 5;
  int lane = threadIdx.x & 31;
  long long n = wave >> 3;
  int head = (int)(wave & 7);
  if (n >= N) return;
  const float* hp = h1 + n * 512 + head * 64;
  const float* as = a1 + head * 128;
  const float* ad = as + 64;
  float ps = 0.f, pd = 0.f;
#pragma unroll
  for (int i = lane; i < 64; i += 32) {
    float hv = hp[i];
    ps += hv * as[i];
    pd += hv * ad[i];
  }
#pragma unroll
  for (int off = 16; off; off >>= 1) {
    ps += __shfl_xor(ps, off, 32);
    pd += __shfl_xor(pd, off, 32);
  }
  if (lane == 0) { ssrc[n * 8 + head] = ps; sdst[n * 8 + head] = pd; }
}

// one wave per node: F=128
__global__ void scores2_kernel(const float* __restrict__ h2,
                               const float* __restrict__ a2,
                               float* __restrict__ ssrc, float* __restrict__ sdst,
                               int N) {
  long long wave = ((long long)blockIdx.x * blockDim.x + threadIdx.x) >> 5;
  int lane = threadIdx.x & 31;
  if (wave >= N) return;
  const float* hp = h2 + wave * 128;
  float ps = 0.f, pd = 0.f;
#pragma unroll
  for (int i = lane; i < 128; i += 32) {
    float hv = hp[i];
    ps += hv * a2[i];
    pd += hv * a2[128 + i];
  }
#pragma unroll
  for (int off = 16; off; off >>= 1) {
    ps += __shfl_xor(ps, off, 32);
    pd += __shfl_xor(pd, off, 32);
  }
  if (lane == 0) { ssrc[wave] = ps; sdst[wave] = pd; }
}

// --------------------------- edge (SpMM) kernels ---------------------------
// one wave per (edge, head); 2 floats/lane of the 64-wide head feature
__global__ void edge1_kernel(const int* __restrict__ src, const int* __restrict__ dst,
                             const float* __restrict__ h1,
                             const float* __restrict__ ssrc, const float* __restrict__ sdst,
                             float* __restrict__ msg, float* __restrict__ rowsum,
                             float* __restrict__ deg, long long E) {
  long long wave = ((long long)blockIdx.x * blockDim.x + threadIdx.x) >> 5;
  int lane = threadIdx.x & 31;
  long long e = wave >> 3;
  int head = (int)(wave & 7);
  if (e >= E) return;
  int s = src[e], d = dst[e];
  float score = ssrc[(size_t)s * 8 + head] + sdst[(size_t)d * 8 + head];
  float lk = score > 0.f ? score : ALPHA * score;
  float w = __expf(-lk);
  float2 hv = reinterpret_cast<const float2*>(h1 + (size_t)d * 512 + head * 64)[lane];
  float* mp = msg + (size_t)s * 512 + head * 64 + lane * 2;
  atomicAdd(mp, w * hv.x);
  atomicAdd(mp + 1, w * hv.y);
  if (lane == 0) {
    atomicAdd(rowsum + (size_t)s * 8 + head, w);
    if (head == 0) atomicAdd(deg + s, 1.0f);
  }
}

// one wave per edge; 4 floats/lane of the 128-wide feature
__global__ void edge2_kernel(const int* __restrict__ src, const int* __restrict__ dst,
                             const float* __restrict__ h2,
                             const float* __restrict__ ssrc, const float* __restrict__ sdst,
                             float* __restrict__ msg, float* __restrict__ rowsum,
                             long long E) {
  long long e = ((long long)blockIdx.x * blockDim.x + threadIdx.x) >> 5;
  int lane = threadIdx.x & 31;
  if (e >= E) return;
  int s = src[e], d = dst[e];
  float score = ssrc[s] + sdst[d];
  float lk = score > 0.f ? score : ALPHA * score;
  float w = __expf(-lk);
  float4 hv = reinterpret_cast<const float4*>(h2 + (size_t)d * 128)[lane];
  float* mp = msg + (size_t)s * 128 + lane * 4;
  atomicAdd(mp + 0, w * hv.x);
  atomicAdd(mp + 1, w * hv.y);
  atomicAdd(mp + 2, w * hv.z);
  atomicAdd(mp + 3, w * hv.w);
  if (lane == 0) atomicAdd(rowsum + s, w);
}

// ----------------------------- epilogues -----------------------------------
// layer1: x_ = elu(msg/rowsum), emitted directly as f16 GEMM input
__global__ void finalize1_kernel(const float* __restrict__ msg,
                                 const float* __restrict__ rowsum,
                                 _Float16* __restrict__ xcat16, long long total) {
  long long i = (long long)blockIdx.x * blockDim.x + threadIdx.x;
  if (i >= total) return;
  long long n = i >> 9;
  int c = (int)(i & 511);
  float v = msg[i] / (rowsum[n * 8 + (c >> 6)] + EPS);
  float e = v > 0.f ? v : (__expf(v) - 1.f);
  xcat16[i] = (_Float16)e;
}

// final: h = msg/rowsum; zero-degree rows fall back to x; out = elu(h)
__global__ void final_kernel(const float* __restrict__ msg,
                             const float* __restrict__ rowsum,
                             const float* __restrict__ deg,
                             const float* __restrict__ x,
                             float* __restrict__ out, long long total) {
  long long i = (long long)blockIdx.x * blockDim.x + threadIdx.x;
  if (i >= total) return;
  long long n = i >> 7;
  float v = msg[i] / (rowsum[n] + EPS);
  float h = (deg[n] == 0.f) ? x[i] : v;
  out[i] = h > 0.f ? h : (__expf(h) - 1.f);
}

// ---------------------------------------------------------------------------
extern "C" void kernel_launch(void* const* d_in, const int* in_sizes, int n_in,
                              void* d_out, int out_size, void* d_ws, size_t ws_size,
                              hipStream_t stream) {
  const float* x  = (const float*)d_in[0];
  const int*   ei = (const int*)d_in[1];
  const float* W1 = (const float*)d_in[2];
  const float* a1 = (const float*)d_in[3];
  const float* W2 = (const float*)d_in[4];
  const float* a2 = (const float*)d_in[5];

  const long long N = in_sizes[0] / 128;   // 100000
  const long long E = in_sizes[1] / 2;     // 3200000
  const int* srcIdx = ei;
  const int* dstIdx = ei + E;

  // ---- workspace carve-out (lifetime aliasing: h1<->msg2, msg1<->h2) ----
  size_t off = 0;
  auto alloc = [&](size_t bytes) -> void* {
    void* p = (char*)d_ws + off;
    off = (off + bytes + 255) & ~(size_t)255;
    return p;
  };
  float*    regionA = (float*)alloc((size_t)N * 512 * 4);  // h1, then msg2
  float*    regionB = (float*)alloc((size_t)N * 512 * 4);  // msg1, then h2
  _Float16* xcat16  = (_Float16*)alloc((size_t)N * 512 * 2);
  _Float16* x16     = (_Float16*)alloc((size_t)N * 128 * 2);
  _Float16* Bt1     = (_Float16*)alloc((size_t)512 * 128 * 2);
  _Float16* Bt2     = (_Float16*)alloc((size_t)128 * 512 * 2);
  float*    s1s     = (float*)alloc((size_t)N * 8 * 4);
  float*    s1d     = (float*)alloc((size_t)N * 8 * 4);
  float*    rs1     = (float*)alloc((size_t)N * 8 * 4);
  float*    s2s     = (float*)alloc((size_t)N * 4);
  float*    s2d     = (float*)alloc((size_t)N * 4);
  float*    rs2     = (float*)alloc((size_t)N * 4);
  float*    deg     = (float*)alloc((size_t)N * 4);

  float* h1   = regionA;
  float* msg2 = regionA;
  float* msg1 = regionB;
  float* h2   = regionB;

  const int B = 256;
  auto blocks = [](long long work, int b) { return (unsigned)((work + b - 1) / b); };

  // 1) convert inputs to f16 (A) and transposed f16 (B)
  conv_f32_f16<<<blocks(N * 128, B), B, 0, stream>>>(x, x16, N * 128);
  conv_W1<<<blocks(512 * 128, B), B, 0, stream>>>(W1, Bt1);
  conv_W2<<<blocks(128 * 512, B), B, 0, stream>>>(W2, Bt2);

  // 2) zero layer-1 accumulators
  hipMemsetAsync(msg1, 0, (size_t)N * 512 * 4, stream);
  hipMemsetAsync(rs1, 0, (size_t)N * 8 * 4, stream);
  hipMemsetAsync(deg, 0, (size_t)N * 4, stream);

  // 3) layer-1 GEMM (all 8 heads fused): h1[N,512] = x16[N,128] @ W1cat
  {
    long long waves = (N + 15) / 16;
    gat_wmma_gemm<<<blocks(waves * 32, B), B, 0, stream>>>(x16, Bt1, h1,
                                                          (int)N, 128, 512);
  }

  // 4) per-(node,head) attention scores
  scores1_kernel<<<blocks((long long)N * 8 * 32, B), B, 0, stream>>>(h1, a1, s1s,
                                                                     s1d, (int)N);

  // 5) layer-1 edge scatter (wave per (edge,head))
  edge1_kernel<<<blocks(E * 8 * 32, B), B, 0, stream>>>(srcIdx, dstIdx, h1, s1s,
                                                        s1d, msg1, rs1, deg, E);

  // 6) layer-1 epilogue -> f16 concat features
  finalize1_kernel<<<blocks(N * 512, B), B, 0, stream>>>(msg1, rs1, xcat16,
                                                         N * 512);

  // 7) zero layer-2 accumulators (aliases regionA; h1 now dead)
  hipMemsetAsync(msg2, 0, (size_t)N * 128 * 4, stream);
  hipMemsetAsync(rs2, 0, (size_t)N * 4, stream);

  // 8) layer-2 GEMM: h2[N,128] = xcat16[N,512] @ W2
  {
    long long waves = (N + 15) / 16;
    gat_wmma_gemm<<<blocks(waves * 32, B), B, 0, stream>>>(xcat16, Bt2, h2,
                                                          (int)N, 512, 128);
  }

  // 9) layer-2 scores
  scores2_kernel<<<blocks((long long)N * 32, B), B, 0, stream>>>(h2, a2, s2s,
                                                                 s2d, (int)N);

  // 10) layer-2 edge scatter (wave per edge)
  edge2_kernel<<<blocks(E * 32, B), B, 0, stream>>>(srcIdx, dstIdx, h2, s2s, s2d,
                                                    msg2, rs2, E);

  // 11) final epilogue: zero-degree fallback + elu
  final_kernel<<<blocks(N * 128, B), B, 0, stream>>>(msg2, rs2, deg, x, d_out ?
                                                     (float*)d_out : nullptr,
                                                     N * 128);
}
